// Atention_47974784697002
// MI455X (gfx1250) — compile-verified, matrix operand order checked by reading, not run on hardware
//
#include <hip/hip_runtime.h>
#include <hip/hip_bf16.h>

typedef __attribute__((ext_vector_type(16))) __bf16 v16bf;
typedef __attribute__((ext_vector_type(8)))  float  v8f;

#define B_    64
#define S_    2048
#define HID_  1024
#define ENC_  2048
#define ATT_  1024
#define MASK_FILL_ -1000000009.0f

// ---------------------------------------------------------------------------
// Kernel 0: pre-convert U_a [ATT,ENC] f32 -> bf16 in WMMA per-lane tile layout.
// slot g = ((nt*64 + kc)*32 + lane); ub[g*16..g*16+15] = the 16 bf16 values
// lane `lane` feeds to WMMA B for N-tile nt, K-chunk kc.
// ---------------------------------------------------------------------------
__global__ __launch_bounds__(256) void k_prep_ua(const float* __restrict__ Ua,
                                                 __bf16* __restrict__ ub) {
  const int g   = blockIdx.x * 256 + threadIdx.x;
  const int ls  = g & 31;
  const int kc  = (g >> 5) & 63;
  const int nt  = g >> 11;
  const int n   = nt * 16 + (ls & 15);
  const int hal = ls >> 4;
  const float* src = Ua + (size_t)n * ENC_ + kc * 32 + hal * 8;
  float4 u0 = *(const float4*)(src);
  float4 u1 = *(const float4*)(src + 4);
  float4 u2 = *(const float4*)(src + 16);
  float4 u3 = *(const float4*)(src + 20);
  v16bf vb;
  vb[0]  = (__bf16)u0.x; vb[1]  = (__bf16)u0.y; vb[2]  = (__bf16)u0.z; vb[3]  = (__bf16)u0.w;
  vb[4]  = (__bf16)u1.x; vb[5]  = (__bf16)u1.y; vb[6]  = (__bf16)u1.z; vb[7]  = (__bf16)u1.w;
  vb[8]  = (__bf16)u2.x; vb[9]  = (__bf16)u2.y; vb[10] = (__bf16)u2.z; vb[11] = (__bf16)u2.w;
  vb[12] = (__bf16)u3.x; vb[13] = (__bf16)u3.y; vb[14] = (__bf16)u3.z; vb[15] = (__bf16)u3.w;
  *(v16bf*)(ub + (size_t)g * 16) = vb;
}

// ---------------------------------------------------------------------------
// Kernel 1: dec_proj[b,a] = sum_h decoder_state[b,h] * W_a[a,h]
// ---------------------------------------------------------------------------
__global__ __launch_bounds__(256) void k_dec_proj(const float* __restrict__ dec,
                                                  const float* __restrict__ Wa,
                                                  float* __restrict__ dp) {
  __shared__ float sdec[HID_];
  const int b = blockIdx.y;
  const int a = blockIdx.x * 256 + threadIdx.x;
  for (int i = threadIdx.x; i < HID_; i += 256) sdec[i] = dec[b * HID_ + i];
  __syncthreads();
  const float4* w = (const float4*)(Wa + (size_t)a * HID_);
  float acc = 0.f;
#pragma unroll 4
  for (int h = 0; h < HID_ / 4; ++h) {
    float4 wv = w[h];
    acc += wv.x * sdec[4 * h + 0] + wv.y * sdec[4 * h + 1] +
           wv.z * sdec[4 * h + 2] + wv.w * sdec[4 * h + 3];
  }
  dp[b * ATT_ + a] = acc;
}

// ---------------------------------------------------------------------------
// CDNA5 hardware tanh (TRANS32 op, co-executes with the matrix pipe)
// ---------------------------------------------------------------------------
__device__ __forceinline__ float fast_tanh(float x) {
  float y;
  asm("v_tanh_f32 %0, %1" : "=v"(y) : "v"(x));
  return y;
}

__device__ __forceinline__ void stage_issue(const __bf16* __restrict__ ub,
                                            unsigned ldsOff, int nb, int kb,
                                            int tid) {
#pragma unroll
  for (int i = 0; i < 16; ++i) {
    const int w    = i * 256 + tid;          // flat b128 index in 64 KB stage
    const int slab = w >> 9;                 // which N-tile slab (8 KB)
    const int win  = w & 511;                // b128 within slab
    const char* ga = (const char*)ub + ((size_t)(nb * 8 + slab) << 16) +
                     ((size_t)kb << 13) + ((size_t)win << 4);
    const unsigned la = ldsOff + ((unsigned)w << 4);
    asm volatile("global_load_async_to_lds_b128 %0, %1, off"
                 :: "v"(la), "v"(ga) : "memory");
  }
}

// ---------------------------------------------------------------------------
// Kernel 2 (hot): fused  E[b,s] = v . tanh(dec_proj[b] + U_a enc[b,s])
// 8 waves x M=16 rows; 8 passes of 128 N-cols (64 f32 acc VGPRs).
// Double-buffered LDS: async DMA (global_load_async_to_lds_b128 / ASYNCcnt)
// of stage kb+1 overlaps the 64 WMMAs of stage kb.
// ---------------------------------------------------------------------------
__global__ __launch_bounds__(256) void k_scores(const float* __restrict__ enc,
                                                const __bf16* __restrict__ ub,
                                                const float* __restrict__ dp,
                                                const float* __restrict__ va,
                                                float* __restrict__ Eij) {
  // [buf][tile][kc][lane][16 bf16] = 2 x 64 KB
  __shared__ __align__(32) __bf16 ldsB[2][8][8][32][16];

  const int tid  = threadIdx.x;
  const int lane = tid & 31;
  const int wave = tid >> 5;
  const int half = lane >> 4;
  const int row  = lane & 15;
  const int b    = blockIdx.y;
  const int s0   = blockIdx.x * 128 + wave * 16;

  const float* Arow = enc + ((size_t)b * S_ + s0 + row) * ENC_ + half * 8;
  const unsigned ldsBase = (unsigned)(size_t)&ldsB[0][0][0][0][0];
  const char* lbase = (const char*)&ldsB[0][0][0][0][0] + lane * 32;

  float rowsum[8];
#pragma unroll
  for (int r = 0; r < 8; ++r) rowsum[r] = 0.f;

  for (int nb = 0; nb < ATT_ / 128; ++nb) {      // 8 passes of 128 N-columns
    v8f acc[8] = {};
    stage_issue(ub, ldsBase, nb, 0, tid);        // prime buffer 0
    asm volatile("s_wait_asynccnt 0" ::: "memory");
    __syncthreads();
    for (int kb = 0; kb < ENC_ / 256; ++kb) {    // K staged in blocks of 256
      const int cur = kb & 1;
      if (kb < 7)                                // DMA next stage into other buf
        stage_issue(ub, ldsBase + ((unsigned)(cur ^ 1) << 16), nb, kb + 1, tid);
      const char* lb = lbase + (cur << 16);
#pragma unroll
      for (int kc = 0; kc < 8; ++kc) {
        // A tile: 16 rows x 32 K, per-lane layout (groups +0/+16 from half*8)
        const float* ap = Arow + kb * 256 + kc * 32;
        float4 a0 = *(const float4*)(ap);
        float4 a1 = *(const float4*)(ap + 4);
        float4 a2 = *(const float4*)(ap + 16);
        float4 a3 = *(const float4*)(ap + 20);
        v16bf Abf;
        Abf[0]  = (__bf16)a0.x; Abf[1]  = (__bf16)a0.y;
        Abf[2]  = (__bf16)a0.z; Abf[3]  = (__bf16)a0.w;
        Abf[4]  = (__bf16)a1.x; Abf[5]  = (__bf16)a1.y;
        Abf[6]  = (__bf16)a1.z; Abf[7]  = (__bf16)a1.w;
        Abf[8]  = (__bf16)a2.x; Abf[9]  = (__bf16)a2.y;
        Abf[10] = (__bf16)a2.z; Abf[11] = (__bf16)a2.w;
        Abf[12] = (__bf16)a3.x; Abf[13] = (__bf16)a3.y;
        Abf[14] = (__bf16)a3.z; Abf[15] = (__bf16)a3.w;
        v16bf Bt[8];
#pragma unroll
        for (int t = 0; t < 8; ++t)
          Bt[t] = *(const v16bf*)(lb + t * 8192 + kc * 1024);
#pragma unroll
        for (int t = 0; t < 8; ++t)
          acc[t] = __builtin_amdgcn_wmma_f32_16x16x32_bf16(
              false, Abf, false, Bt[t], (short)0, acc[t], false, false);
      }
      if (kb < 7) {                              // next buffer landed + all
        asm volatile("s_wait_asynccnt 0" ::: "memory"); // waves done with cur
        __syncthreads();
      }
    }
    // epilogue: tanh(enc_proj + dec_proj) * v_a, reduce over these 128 cols
#pragma unroll
    for (int t = 0; t < 8; ++t) {
      const int n = nb * 128 + t * 16 + row;
      const float d  = dp[b * ATT_ + n];
      const float vv = va[n];
#pragma unroll
      for (int r = 0; r < 8; ++r)
        rowsum[r] += fast_tanh(acc[t][r] + d) * vv;
    }
  }
  // cross-lane reduce: C layout holds row r in lanes 0-15, row r+8 in 16-31
#pragma unroll
  for (int r = 0; r < 8; ++r) {
    float v = rowsum[r];
    v += __shfl_xor(v, 1, 32);
    v += __shfl_xor(v, 2, 32);
    v += __shfl_xor(v, 4, 32);
    v += __shfl_xor(v, 8, 32);
    if (row == 0) Eij[b * S_ + s0 + r + half * 8] = v;
  }
}

// ---------------------------------------------------------------------------
// Kernel 3: masked softmax over S per batch row -> alpha
// ---------------------------------------------------------------------------
__global__ __launch_bounds__(256) void k_softmax(const float* __restrict__ Eij,
                                                 const int* __restrict__ mask,
                                                 float* __restrict__ alpha) {
  __shared__ float red[256];
  const int b = blockIdx.x, tid = threadIdx.x;
  float m = -3.4e38f;
  for (int s = tid; s < S_; s += 256) {
    float e = (mask[b * S_ + s] == 0) ? MASK_FILL_ : Eij[b * S_ + s];
    m = fmaxf(m, e);
  }
  red[tid] = m;
  __syncthreads();
  for (int o = 128; o > 0; o >>= 1) {
    if (tid < o) red[tid] = fmaxf(red[tid], red[tid + o]);
    __syncthreads();
  }
  m = red[0];
  __syncthreads();
  float sum = 0.f;
  for (int s = tid; s < S_; s += 256) {
    float e = (mask[b * S_ + s] == 0) ? MASK_FILL_ : Eij[b * S_ + s];
    float ex = expf(e - m);
    alpha[b * S_ + s] = ex;
    sum += ex;
  }
  red[tid] = sum;
  __syncthreads();
  for (int o = 128; o > 0; o >>= 1) {
    if (tid < o) red[tid] += red[tid + o];
    __syncthreads();
  }
  const float inv = 1.f / red[0];
  for (int s = tid; s < S_; s += 256) alpha[b * S_ + s] *= inv;
}

// ---------------------------------------------------------------------------
// Kernel 4: context[b,e] = sum_s alpha[b,s] * enc[b,s,e]  (memory bound)
// ---------------------------------------------------------------------------
__global__ __launch_bounds__(256) void k_context(const float* __restrict__ enc,
                                                 const float* __restrict__ alpha,
                                                 float* __restrict__ ctx) {
  __shared__ float sal[256];
  const int b = blockIdx.y;
  const int e = blockIdx.x * 256 + threadIdx.x;
  float acc = 0.f;
  for (int sb = 0; sb < S_; sb += 256) {
    __syncthreads();
    sal[threadIdx.x] = alpha[b * S_ + sb + threadIdx.x];
    __syncthreads();
    const float* ep = enc + ((size_t)b * S_ + sb) * ENC_ + e;
#pragma unroll 8
    for (int j = 0; j < 256; ++j) acc += sal[j] * ep[(size_t)j * ENC_];
  }
  ctx[b * ENC_ + e] = acc;
}

// ---------------------------------------------------------------------------
extern "C" void kernel_launch(void* const* d_in, const int* in_sizes, int n_in,
                              void* d_out, int out_size, void* d_ws, size_t ws_size,
                              hipStream_t stream) {
  const float* dec  = (const float*)d_in[0];   // [B,HID]
  const float* enc  = (const float*)d_in[1];   // [B,S,ENC]
  const int*   msk  = (const int*)  d_in[2];   // [B,S]
  const float* Wa   = (const float*)d_in[3];   // [ATT,HID]
  const float* Ua   = (const float*)d_in[4];   // [ATT,ENC]
  const float* va   = (const float*)d_in[5];   // [ATT]

  __bf16* ubf  = (__bf16*)d_ws;                          // [ATT*ENC] bf16, 4 MB
  float*  dpw  = (float*)((char*)d_ws + (size_t)ATT_ * ENC_ * 2); // [B,ATT]
  float*  Eij  = dpw + (size_t)B_ * ATT_;                // [B,S]
  float*  ctx  = (float*)d_out;                          // [B,ENC]
  float*  alpha = ctx + (size_t)B_ * ENC_;               // [B,S]

  k_prep_ua <<<(ATT_ / 16) * (ENC_ / 32) * 32 / 256, 256, 0, stream>>>(Ua, ubf);
  k_dec_proj<<<dim3(ATT_ / 256, B_), 256, 0, stream>>>(dec, Wa, dpw);
  k_scores  <<<dim3(S_ / 128, B_),   256, 0, stream>>>(enc, ubf, dpw, va, Eij);
  k_softmax <<<B_,                   256, 0, stream>>>(Eij, msk, alpha);
  k_context <<<dim3(ENC_ / 256, B_), 256, 0, stream>>>(enc, alpha, ctx);
}
